// PointNetSetAbstraction_70291434766903
// MI455X (gfx1250) — matrix operand order repaired and checked by reading, not compiled, not run on hardware
//
#include <hip/hip_runtime.h>
#include <hip/hip_bf16.h>

// ---------------- problem constants ----------------
#define BB 16
#define NN 8192
#define SS 2048
#define KK 32          // nsample
#define DF 64          // input feature channels
#define C0 67          // 3 + DF
#define K0P 96         // padded K for layer0 (multiple of 32)
#define RTOT (BB * SS * KK)   // 1,048,576 rows through the MLP
#define RAD2 0.2025f   // RADIUS**2 (compared against *euclidean distance*, per reference)

typedef __attribute__((ext_vector_type(16))) __bf16 v16bf;
typedef __attribute__((ext_vector_type(8)))  float  v8f;
typedef __attribute__((ext_vector_type(4))) unsigned int u32x4;
typedef __attribute__((ext_vector_type(8))) int          i32x8;
typedef __attribute__((ext_vector_type(4))) int          i32x4;

#if __has_builtin(__builtin_amdgcn_tensor_load_to_lds) && __has_builtin(__builtin_amdgcn_s_wait_tensorcnt)
#define HAVE_TDM 1
#else
#define HAVE_TDM 0
#endif

// ============================================================
// 1) transposes:  xyz [B,3,N] -> xyz_t [B,N,3] ;  points [B,64,N] -> pts_t [B,N,64]
// ============================================================
__global__ void transpose_xyz_kernel(const float* __restrict__ xyz, float* __restrict__ xyz_t) {
    int i = blockIdx.x * 256 + threadIdx.x;           // b*N + n
    if (i < BB * NN) {
        int b = i >> 13, n = i & (NN - 1);
#pragma unroll
        for (int c = 0; c < 3; ++c)
            xyz_t[(long)i * 3 + c] = xyz[((long)b * 3 + c) * NN + n];
    }
}

__global__ void transpose_pts_kernel(const float* __restrict__ pts, float* __restrict__ pts_t) {
    __shared__ float tile[64][65];
    int b = blockIdx.y;
    int n0 = blockIdx.x * 64;
    int tx = threadIdx.x & 63, ty = threadIdx.x >> 6;     // 256 threads
    for (int c = ty; c < 64; c += 4)
        tile[c][tx] = pts[((long)b * 64 + c) * NN + n0 + tx];
    __syncthreads();
    for (int n = ty; n < 64; n += 4)
        pts_t[((long)b * NN + n0 + n) * 64 + tx] = tile[tx][n];
}

// ============================================================
// 2) Farthest point sampling: 1 workgroup per batch, LDS-resident.
// ============================================================
__global__ __launch_bounds__(1024) void fps_kernel(const float* __restrict__ xyz,
                                                   float* __restrict__ new_xyz_t,
                                                   float* __restrict__ out_newxyz) {
    extern __shared__ float fsm[];
    float* px = fsm;            // [8192]
    float* py = px + NN;
    float* pz = py + NN;
    float* dist = pz + NN;
    __shared__ float rv[32];
    __shared__ int   ri[32];
    __shared__ int   s_far;

    const int b = blockIdx.x;
    const int t = threadIdx.x;

    for (int j = t; j < NN; j += 1024) {
        px[j]   = xyz[((long)b * 3 + 0) * NN + j];
        py[j]   = xyz[((long)b * 3 + 1) * NN + j];
        pz[j]   = xyz[((long)b * 3 + 2) * NN + j];
        dist[j] = 1e10f;
    }
    __syncthreads();

    int far = 0;
    for (int i = 0; i < SS; ++i) {
        float cx = px[far], cy = py[far], cz = pz[far];
        if (t == 0) {
            new_xyz_t[((long)b * SS + i) * 3 + 0] = cx;
            new_xyz_t[((long)b * SS + i) * 3 + 1] = cy;
            new_xyz_t[((long)b * SS + i) * 3 + 2] = cz;
            out_newxyz[((long)b * 3 + 0) * SS + i] = cx;
            out_newxyz[((long)b * 3 + 1) * SS + i] = cy;
            out_newxyz[((long)b * 3 + 2) * SS + i] = cz;
        }
        float bv = -1.0f; int bi = 0x7fffffff;
        for (int j = t; j < NN; j += 1024) {
            float dx = px[j] - cx, dy = py[j] - cy, dz = pz[j] - cz;
            float d = dx * dx + dy * dy + dz * dz;
            float dm = fminf(dist[j], d);
            dist[j] = dm;
            if (dm > bv || (dm == bv && j < bi)) { bv = dm; bi = j; }
        }
#pragma unroll
        for (int off = 16; off > 0; off >>= 1) {
            float ov = __shfl_down(bv, off, 32);
            int   oi = __shfl_down(bi, off, 32);
            if (ov > bv || (ov == bv && oi < bi)) { bv = ov; bi = oi; }
        }
        if ((t & 31) == 0) { rv[t >> 5] = bv; ri[t >> 5] = bi; }
        __syncthreads();
        if (t == 0) {
            float fv = rv[0]; int fi = ri[0];
            for (int q = 1; q < 32; ++q)
                if (rv[q] > fv || (rv[q] == fv && ri[q] < fi)) { fv = rv[q]; fi = ri[q]; }
            s_far = fi;
        }
        __syncthreads();
        far = s_far;
        __syncthreads();
    }
}

// ============================================================
// 3) Ball query: block per centroid, d^2 in registers, 32 argmin rounds
// ============================================================
__global__ __launch_bounds__(256) void ballquery_kernel(const float* __restrict__ xyz_t,
                                                        const float* __restrict__ new_xyz_t,
                                                        int* __restrict__ gidx) {
    const int bs = blockIdx.x;            // b*S + s
    const int b = bs >> 11;
    const int t = threadIdx.x;            // 256
    const float cx = new_xyz_t[(long)bs * 3 + 0];
    const float cy = new_xyz_t[(long)bs * 3 + 1];
    const float cz = new_xyz_t[(long)bs * 3 + 2];

    float d2[32];
#pragma unroll
    for (int u = 0; u < 32; ++u) {
        int j = u * 256 + t;
        const float* p = xyz_t + ((long)b * NN + j) * 3;
        float dx = p[0] - cx, dy = p[1] - cy, dz = p[2] - cz;
        d2[u] = dx * dx + dy * dy + dz * dz;
    }

    __shared__ float rv[8];
    __shared__ int   ri[8];
    __shared__ int   s_bi;

    for (int sel = 0; sel < KK; ++sel) {
        float bv = 3.4e38f; int bi = 0x7fffffff;
#pragma unroll
        for (int u = 0; u < 32; ++u) {
            int j = u * 256 + t;
            float v = d2[u];
            if (v < bv || (v == bv && j < bi)) { bv = v; bi = j; }
        }
#pragma unroll
        for (int off = 16; off > 0; off >>= 1) {
            float ov = __shfl_down(bv, off, 32);
            int   oi = __shfl_down(bi, off, 32);
            if (ov < bv || (ov == bv && oi < bi)) { bv = ov; bi = oi; }
        }
        if ((t & 31) == 0) { rv[t >> 5] = bv; ri[t >> 5] = bi; }
        __syncthreads();
        if (t == 0) {
            float fv = rv[0]; int fi = ri[0];
            for (int q = 1; q < 8; ++q)
                if (rv[q] < fv || (rv[q] == fv && ri[q] < fi)) { fv = rv[q]; fi = ri[q]; }
            s_bi = fi;
            gidx[(long)bs * KK + sel] = (sqrtf(fmaxf(fv, 0.f)) < RAD2) ? fi : 0;
        }
        __syncthreads();
        int win = s_bi;
        if (t == (win & 255)) d2[win >> 8] = 3.4e38f;
        __syncthreads();
    }
}

// ============================================================
// 4) build layer-0 activations: A0 bf16 [R, 96]
// ============================================================
__global__ __launch_bounds__(128) void build_feats_kernel(const float* __restrict__ xyz_t,
                                                          const float* __restrict__ pts_t,
                                                          const float* __restrict__ new_xyz_t,
                                                          const int* __restrict__ gidx,
                                                          __bf16* __restrict__ A0) {
    const long r = blockIdx.x;            // row = ((b*S)+s)*K + k
    const int  c = threadIdx.x;           // 0..127, active < 96
    if (c >= K0P) return;
    const long bs = r >> 5;               // b*S + s
    const int  b  = (int)(bs >> 11);
    const int  idx = gidx[r];
    float v = 0.f;
    if (c < 3) {
        v = xyz_t[((long)b * NN + idx) * 3 + c] - new_xyz_t[bs * 3 + c];
    } else if (c < C0) {
        v = pts_t[((long)b * NN + idx) * 64 + (c - 3)];
    }
    A0[r * K0P + c] = (__bf16)v;
}

// ============================================================
// 5) weight prep: f32 [O,Cin] -> bf16 [O,Kpad] zero padded
// ============================================================
__global__ void prep_w_kernel(const float* __restrict__ W, __bf16* __restrict__ Wp,
                              int O, int Cin, int Kpad) {
    int i = blockIdx.x * 256 + threadIdx.x;
    if (i < O * Kpad) {
        int o = i / Kpad, k = i % Kpad;
        Wp[i] = (__bf16)(k < Cin ? W[o * Cin + k] : 0.f);
    }
}

// ============================================================
// 6) WMMA GEMM with TDM-staged weights in LDS.
//    H[r,o] = sum_k A[r,k]*W[o,k] + bias[o]; accumulates BN sum/sumsq.
//    Block = 8 waves; each wave owns one 16-row tile and all O columns.
// ============================================================
template <int K, int O>
__global__ __launch_bounds__(256) void mlp_gemm_kernel(const __bf16* __restrict__ A,
                                                       const __bf16* __restrict__ W,
                                                       const float* __restrict__ bias,
                                                       __bf16* __restrict__ H,
                                                       float* __restrict__ gsum,
                                                       float* __restrict__ gsq) {
    constexpr int CH = K / 32;    // k-chunks
    constexpr int CT = O / 16;    // col tiles
    __shared__ __align__(16) __bf16 lw[O * K];   // TDM-staged weights
    __shared__ float lsum[O], lsq[O];
    const int t = threadIdx.x;
    if (t < O) { lsum[t] = 0.f; lsq[t] = 0.f; }

#if HAVE_TDM
    // ---- Tensor Data Mover: DMA the whole [O x K] bf16 weight matrix into LDS.
    // D# per ISA 8.3/8.4: 2D tensor, data_size=2B, dim0=K, dim1=O, tile==tensor.
    if (t < 32) {
        unsigned lds_off = (unsigned)(unsigned long long)(uintptr_t)&lw[0];
        unsigned long long ga = (unsigned long long)(uintptr_t)W;
        u32x4 g0 = { 1u,                                       // count=1 (valid), no gather
                     lds_off,                                  // lds_addr
                     (unsigned)(ga & 0xffffffffu),             // global_addr[31:0]
                     (unsigned)((ga >> 32) & 0x01ffffffu) | (2u << 30) }; // addr[56:32] | type=2
        i32x8 g1 = { (int)(1u << 16),                          // data_size=1 (2 bytes)
                     (int)((K & 0xffff) << 16),                // tensor_dim0 lo16 (bits 63:48)
                     (int)(((unsigned)K >> 16) | ((O & 0xffff) << 16)),   // dim0 hi | dim1 lo
                     (int)(((unsigned)O >> 16) | ((K & 0xffff) << 16)),   // dim1 hi | tile_dim0=K
                     (int)(O & 0xffff),                        // tile_dim1=O, tile_dim2=0
                     (int)K,                                   // tensor_dim0_stride lo32
                     0, 0 };                                   // stride hi / dim1_stride = 0
        i32x4 gz  = { 0, 0, 0, 0 };                            // groups 2/3: unused (2D tensor)
        i32x8 gz8 = { 0, 0, 0, 0, 0, 0, 0, 0 };                // trailing group (clang-23 6-arg form)
        __builtin_amdgcn_tensor_load_to_lds(g0, g1, gz, gz, gz8, 0);
    }
#else
    for (int i = t; i < O * K / 8; i += 256)
        ((uint4*)lw)[i] = ((const uint4*)W)[i];
#endif

    const int lane = t & 31;
    const int wave = t >> 5;
    const long m0 = ((long)blockIdx.x * 8 + wave) * 16;

    // ---- A fragments (ISA 7.12.2: lane<16 -> K {c*32+0..7, c*32+16..23}) ----
    // Overlaps with the in-flight TDM transfer.
    const int rA  = (int)m0 + (lane & 15);
    const int kbA = (lane & 16) ? 8 : 0;
    v16bf afrag[CH];
#pragma unroll
    for (int c = 0; c < CH; ++c) {
        const __bf16* p = A + (long)rA * K + c * 32 + kbA;
        union { v16bf v; uint4 q[2]; } u;
        u.q[0] = *(const uint4*)(p);
        u.q[1] = *(const uint4*)(p + 16);
        afrag[c] = u.v;
    }
    // hint the next block's A rows toward the caches (global_prefetch_b8)
    __builtin_prefetch(A + ((long)rA + 128) * K, 0, 1);

#if HAVE_TDM
    if (t < 32) __builtin_amdgcn_s_wait_tensorcnt(0);
#endif
    __syncthreads();   // weights visible in LDS to all waves

    const int nB  = lane & 15;
    const int kbB = (lane & 16) ? 8 : 0;
#pragma unroll
    for (int ot = 0; ot < CT; ++ot) {
        const int o0 = ot * 16;
        v8f acc = {};
#pragma unroll
        for (int c = 0; c < CH; ++c) {
            const __bf16* p = lw + (o0 + nB) * K + c * 32 + kbB;   // ds_load_b128 x2
            union { v16bf v; uint4 q[2]; } u;
            u.q[0] = *(const uint4*)(p);
            u.q[1] = *(const uint4*)(p + 16);
            acc = __builtin_amdgcn_wmma_f32_16x16x32_bf16(
                false, afrag[c], false, u.v, (short)0, acc, false, false);
        }
        // epilogue: lane<16: col=o0+lane, rows m0+v ; lane>=16: rows m0+8+v
        const int col   = o0 + nB;
        const float bz  = bias[col];
        const int rbase = (int)m0 + ((lane & 16) ? 8 : 0);
        float s = 0.f, q = 0.f;
#pragma unroll
        for (int v = 0; v < 8; ++v) {
            float h = acc[v] + bz;
            H[(long)(rbase + v) * O + col] = (__bf16)h;
            s += h; q += h * h;
        }
        atomicAdd(&lsum[col], s);
        atomicAdd(&lsq[col], q);
    }
    __syncthreads();
    if (t < O) { atomicAdd(&gsum[t], lsum[t]); atomicAdd(&gsq[t], lsq[t]); }
}

// ============================================================
// 7) BN finalize + vectorized BN/ReLU (in-place on bf16 activations)
// ============================================================
__global__ void bn_finalize_kernel(const float* __restrict__ sums, const float* __restrict__ sumsq,
                                   const float* __restrict__ gamma, const float* __restrict__ beta,
                                   float* __restrict__ scale, float* __restrict__ shift,
                                   int C, float invCount) {
    int c = blockIdx.x * 128 + threadIdx.x;
    if (c < C) {
        float m = sums[c] * invCount;
        float v = sumsq[c] * invCount - m * m;
        float sc = gamma[c] * rsqrtf(v + 1e-5f);
        scale[c] = sc;
        shift[c] = beta[c] - m * sc;
    }
}

template <int O>
__global__ void bn_relu_kernel(__bf16* __restrict__ h, const float* __restrict__ scale,
                               const float* __restrict__ shift) {
    // 8 bf16 per thread (one 16-byte vector); O is a multiple of 8.
    long i8 = ((long)blockIdx.x * 256 + threadIdx.x) * 8;
    if (i8 < (long)RTOT * O) {
        union { uint4 q; __bf16 e[8]; } u;
        u.q = *(uint4*)(h + i8);
        int c0 = (int)(i8 % O);
#pragma unroll
        for (int j = 0; j < 8; ++j) {
            float v = (float)u.e[j] * scale[c0 + j] + shift[c0 + j];
            u.e[j] = (__bf16)fmaxf(v, 0.f);
        }
        *(uint4*)(h + i8) = u.q;
    }
}

// ============================================================
// 8) fused BN + ReLU + max over nsample -> new_points [B,128,S]
// ============================================================
__global__ __launch_bounds__(128) void maxpool_out_kernel(const __bf16* __restrict__ h2,
                                                          const float* __restrict__ scale,
                                                          const float* __restrict__ shift,
                                                          float* __restrict__ outp) {
    const int bs = blockIdx.x;          // b*S + s
    const int o  = threadIdx.x;         // 0..127
    const int b  = bs >> 11;
    const int s  = bs & (SS - 1);
    const float sc = scale[o], sh = shift[o];
    const __bf16* p = h2 + (long)bs * KK * 128 + o;
    float m = -3.4e38f;
#pragma unroll
    for (int k = 0; k < KK; ++k) {
        float v = fmaxf((float)p[(long)k * 128] * sc + sh, 0.f);
        m = fmaxf(m, v);
    }
    outp[((long)b * 128 + o) * SS + s] = m;
}

// ============================================================
// host-side launcher
// ============================================================
extern "C" void kernel_launch(void* const* d_in, const int* in_sizes, int n_in,
                              void* d_out, int out_size, void* d_ws, size_t ws_size,
                              hipStream_t stream) {
    const float* xyz    = (const float*)d_in[0];
    const float* points = (const float*)d_in[1];
    const float* W0 = (const float*)d_in[2];
    const float* b0 = (const float*)d_in[3];
    const float* g0 = (const float*)d_in[4];
    const float* be0 = (const float*)d_in[5];
    const float* W1 = (const float*)d_in[6];
    const float* b1 = (const float*)d_in[7];
    const float* g1 = (const float*)d_in[8];
    const float* be1 = (const float*)d_in[9];
    const float* W2 = (const float*)d_in[10];
    const float* b2 = (const float*)d_in[11];
    const float* g2 = (const float*)d_in[12];
    const float* be2 = (const float*)d_in[13];

    float* out_newxyz = (float*)d_out;                       // [16,3,2048]
    float* out_newpts = (float*)d_out + (long)BB * 3 * SS;   // [16,128,2048]

    // ---- workspace carve-out ----
    char* ws = (char*)d_ws;
    size_t off = 0;
    auto take = [&](size_t bytes) { size_t o = off; off = (off + bytes + 255) & ~(size_t)255; return o; };
    float*  xyz_t     = (float*)(ws + take((size_t)BB * NN * 3 * 4));
    float*  pts_t     = (float*)(ws + take((size_t)BB * NN * 64 * 4));
    float*  new_xyz_t = (float*)(ws + take((size_t)BB * SS * 3 * 4));
    int*    gidx      = (int*)  (ws + take((size_t)BB * SS * KK * 4));
    float*  st        = (float*)(ws + take(1024 * 4));       // stats block
    __bf16* Wp0       = (__bf16*)(ws + take((size_t)64 * K0P * 2));
    __bf16* Wp1       = (__bf16*)(ws + take((size_t)64 * 64 * 2));
    __bf16* Wp2       = (__bf16*)(ws + take((size_t)128 * 64 * 2));
    __bf16* A0        = (__bf16*)(ws + take((size_t)RTOT * K0P * 2));
    __bf16* H0        = (__bf16*)(ws + take((size_t)RTOT * 64 * 2));
    __bf16* H1        = (__bf16*)(ws + take((size_t)RTOT * 64 * 2));
    __bf16* H2        = (__bf16*)(ws + take((size_t)RTOT * 128 * 2));
    (void)ws_size; (void)in_sizes; (void)n_in; (void)out_size;

    float* sum0 = st + 0,   *sq0 = st + 64;
    float* sum1 = st + 128, *sq1 = st + 192;
    float* sum2 = st + 256, *sq2 = st + 384;
    float* sc0 = st + 512, *sh0 = st + 576;
    float* sc1 = st + 640, *sh1 = st + 704;
    float* sc2 = st + 768, *sh2 = st + 896;

    (void)hipMemsetAsync(st, 0, 512 * sizeof(float), stream);

    transpose_xyz_kernel<<<(BB * NN + 255) / 256, 256, 0, stream>>>(xyz, xyz_t);
    transpose_pts_kernel<<<dim3(NN / 64, BB), 256, 0, stream>>>(points, pts_t);

    fps_kernel<<<BB, 1024, 4 * NN * sizeof(float), stream>>>(xyz, new_xyz_t, out_newxyz);

    ballquery_kernel<<<BB * SS, 256, 0, stream>>>(xyz_t, new_xyz_t, gidx);

    build_feats_kernel<<<RTOT, 128, 0, stream>>>(xyz_t, pts_t, new_xyz_t, gidx, A0);

    prep_w_kernel<<<(64 * K0P + 255) / 256, 256, 0, stream>>>(W0, Wp0, 64, C0, K0P);
    prep_w_kernel<<<(64 * 64 + 255) / 256, 256, 0, stream>>>(W1, Wp1, 64, 64, 64);
    prep_w_kernel<<<(128 * 64 + 255) / 256, 256, 0, stream>>>(W2, Wp2, 128, 64, 64);

    const float invCnt = 1.0f / (float)RTOT;

    // ---- layer 0: [R,96] x [96,64] ----
    mlp_gemm_kernel<K0P, 64><<<RTOT / 128, 256, 0, stream>>>(A0, Wp0, b0, H0, sum0, sq0);
    bn_finalize_kernel<<<1, 128, 0, stream>>>(sum0, sq0, g0, be0, sc0, sh0, 64, invCnt);
    bn_relu_kernel<64><<<(int)(((long)RTOT * 64 / 8 + 255) / 256), 256, 0, stream>>>(H0, sc0, sh0);

    // ---- layer 1: [R,64] x [64,64] ----
    mlp_gemm_kernel<64, 64><<<RTOT / 128, 256, 0, stream>>>(H0, Wp1, b1, H1, sum1, sq1);
    bn_finalize_kernel<<<1, 128, 0, stream>>>(sum1, sq1, g1, be1, sc1, sh1, 64, invCnt);
    bn_relu_kernel<64><<<(int)(((long)RTOT * 64 / 8 + 255) / 256), 256, 0, stream>>>(H1, sc1, sh1);

    // ---- layer 2: [R,64] x [64,128] ----
    mlp_gemm_kernel<64, 128><<<RTOT / 128, 256, 0, stream>>>(H1, Wp2, b2, H2, sum2, sq2);
    bn_finalize_kernel<<<1, 128, 0, stream>>>(sum2, sq2, g2, be2, sc2, sh2, 128, invCnt);

    maxpool_out_kernel<<<BB * SS, 128, 0, stream>>>(H2, sc2, sh2, out_newpts);
}
